// PhaseAttentionBlock_31001073943257
// MI455X (gfx1250) — matrix operand order repaired and checked by reading, not compiled, not run on hardware
//
#include <hip/hip_runtime.h>

// ---------------------------------------------------------------------------
// Complex phase-attention block on gfx1250 (MI455X), bf16 WMMA pipeline.
// B=2 N=2048 D=1024 H=16 HD=64.  All GEMMs via v_wmma_f32_16x16x32_bf16.
// Attention K/V staged to LDS with global_load_async_to_lds_b128 (ASYNCcnt),
// double buffered across 32-key blocks.
// ---------------------------------------------------------------------------

#define B_  2
#define N_  2048
#define D_  1024
#define H_  16
#define HD_ 64
#define BN_ (B_*N_)
#define SCALE_ 0.125f
#define LOG2E_ 1.4426950408889634f

static const size_t S_  = (size_t)BN_ * D_;   // 4,194,304 elems
static const size_t W_  = (size_t)D_ * D_;    // 1,048,576 elems

typedef __bf16 bf16_t;
typedef __bf16 v8bf  __attribute__((ext_vector_type(8)));
typedef __bf16 v16bf __attribute__((ext_vector_type(16)));
typedef float  v8f   __attribute__((ext_vector_type(8)));

// ---- WMMA helpers ---------------------------------------------------------
__device__ __forceinline__ v8f wmma_bf16(v16bf a, v16bf b, v8f c) {
  return __builtin_amdgcn_wmma_f32_16x16x32_bf16(false, a, false, b,
                                                 (short)0, c, false, false);
}
__device__ __forceinline__ v8f vzero() {
  return (v8f){0.f,0.f,0.f,0.f,0.f,0.f,0.f,0.f};
}

// A-matrix 16x32 bf16 tile (lane = row M=lane%16).
// lane<16 : K = {0..7, 16..23}; lane>=16 : K = {8..15, 24..31}.
__device__ __forceinline__ v16bf load_A(const bf16_t* rowbase, bool hi) {
  const int o1 = hi ? 8 : 0;
  v8bf a = *reinterpret_cast<const v8bf*>(rowbase + o1);
  v8bf b = *reinterpret_cast<const v8bf*>(rowbase + o1 + 16);
  v16bf r;
#pragma unroll
  for (int i = 0; i < 8; ++i) { r[i] = a[i]; r[8 + i] = b[i]; }
  return r;
}

// B-matrix 32x16 bf16 tile (lane = column N=lane%16).
// lane<16 : K = {0..7, 16..23}; lane>=16 : K = {8..15, 24..31}.
__device__ __forceinline__ v16bf load_B(const bf16_t* colbase, bool hi) {
  const int o = hi ? 8 : 0;
  v8bf a = *reinterpret_cast<const v8bf*>(colbase + o);
  v8bf b = *reinterpret_cast<const v8bf*>(colbase + o + 16);
  v16bf r;
#pragma unroll
  for (int i = 0; i < 8; ++i) { r[i] = a[i]; r[8 + i] = b[i]; }
  return r;
}

// ---- fp32 -> bf16 conversion ---------------------------------------------
__global__ __launch_bounds__(256) void k_convert(const float* __restrict__ src,
                                                 bf16_t* __restrict__ dst,
                                                 int n) {
  int i = blockIdx.x * blockDim.x + threadIdx.x;
  int stride = gridDim.x * blockDim.x;
  for (; i < n; i += stride) dst[i] = (bf16_t)src[i];
}

// ---- fused QKV complex projection ----------------------------------------
// accR = Zr@Wr^T - Zi@Wi^T + br ; accI = Zr@Wi^T + Zi@Wr^T + bi
// Dual accumulator chains avoid negated operand copies. 16x32 tile per wave.
// Q scaled by 1/sqrt(HD); V stored head-transposed for attention.
__global__ __launch_bounds__(256) void k_qkv(
    const bf16_t* __restrict__ Zr, const bf16_t* __restrict__ Zi,
    const bf16_t* __restrict__ wqr, const bf16_t* __restrict__ wqi,
    const bf16_t* __restrict__ wkr, const bf16_t* __restrict__ wki,
    const bf16_t* __restrict__ wvr, const bf16_t* __restrict__ wvi,
    const float* __restrict__ bqr, const float* __restrict__ bqi,
    const float* __restrict__ bkr, const float* __restrict__ bki,
    const float* __restrict__ bvr, const float* __restrict__ bvi,
    bf16_t* __restrict__ Qr, bf16_t* __restrict__ Qi,
    bf16_t* __restrict__ Kr, bf16_t* __restrict__ Ki,
    bf16_t* __restrict__ Vtr, bf16_t* __restrict__ Vti) {
  const int wave = (blockIdx.x * blockDim.x + threadIdx.x) >> 5;
  const int lane = threadIdx.x & 31;
  const bool hi = lane >= 16;
  const int l16 = lane & 15;

  const int p    = wave / (256 * 32);      // 0=Q 1=K 2=V
  const int rem  = wave % (256 * 32);
  const int rowT = rem & 255;              // BN/16 row tiles (fast -> B reuse)
  const int colT = rem >> 8;               // D/32 col super-tiles
  const int n0 = rowT * 16, j0 = colT * 32;

  const bf16_t* wr; const bf16_t* wi;
  if (p == 0)      { wr = wqr; wi = wqi; }
  else if (p == 1) { wr = wkr; wi = wki; }
  else             { wr = wvr; wi = wvi; }

  const size_t arow  = (size_t)(n0 + l16) * D_;
  const size_t brow0 = (size_t)(j0 + l16) * D_;
  const size_t brow1 = (size_t)(j0 + 16 + l16) * D_;

  v8f c1[2], c2[2], c3[2], c4[2];
#pragma unroll
  for (int u = 0; u < 2; ++u) { c1[u]=vzero(); c2[u]=vzero(); c3[u]=vzero(); c4[u]=vzero(); }

  for (int k0 = 0; k0 < D_; k0 += 32) {
    v16bf azr = load_A(Zr + arow + k0, hi);
    v16bf azi = load_A(Zi + arow + k0, hi);
#pragma unroll
    for (int u = 0; u < 2; ++u) {
      const size_t brow = u ? brow1 : brow0;
      v16bf bwr = load_B(wr + brow + k0, hi);
      v16bf bwi = load_B(wi + brow + k0, hi);
      c1[u] = wmma_bf16(azr, bwr, c1[u]);
      c2[u] = wmma_bf16(azi, bwi, c2[u]);
      c3[u] = wmma_bf16(azr, bwi, c3[u]);
      c4[u] = wmma_bf16(azi, bwr, c4[u]);
    }
  }

#pragma unroll
  for (int u = 0; u < 2; ++u) {
    const int d = j0 + u * 16 + l16;
    const v8f accR = c1[u] - c2[u];
    const v8f accI = c3[u] + c4[u];
    if (p == 0) {
      const float br = bqr[d], bi = bqi[d];
#pragma unroll
      for (int r = 0; r < 8; ++r) {
        const int n = n0 + r + (hi ? 8 : 0);
        Qr[(size_t)n * D_ + d] = (bf16_t)((accR[r] + br) * SCALE_);
        Qi[(size_t)n * D_ + d] = (bf16_t)((accI[r] + bi) * SCALE_);
      }
    } else if (p == 1) {
      const float br = bkr[d], bi = bki[d];
#pragma unroll
      for (int r = 0; r < 8; ++r) {
        const int n = n0 + r + (hi ? 8 : 0);
        Kr[(size_t)n * D_ + d] = (bf16_t)(accR[r] + br);
        Ki[(size_t)n * D_ + d] = (bf16_t)(accI[r] + bi);
      }
    } else {
      const float br = bvr[d], bi = bvi[d];
      const int h = d >> 6, hd = d & 63;
#pragma unroll
      for (int r = 0; r < 8; ++r) {
        const int ng = n0 + r + (hi ? 8 : 0);
        const int b = ng >> 11, n = ng & (N_ - 1);
        const size_t o = ((size_t)(b * H_ + h) * HD_ + hd) * (size_t)N_ + n;
        Vtr[o] = (bf16_t)(accR[r] + br);
        Vti[o] = (bf16_t)(accI[r] + bi);
      }
    }
  }
}

// ---- flash attention with async K/V staging -------------------------------
// Block = 8 waves = 128 query rows of one (b,h). K/V 32-key blocks staged to
// LDS via global_load_async_to_lds_b128, double buffered.
// scores = Qr@Kr^T - Qi@Ki^T (scale folded into Q), dual accumulators.
#define KSTR 72   // LDS row stride (bf16) for K tiles: 32 keys x 64 hd
#define VSTR 40   // LDS row stride (bf16) for V tiles: 64 hd  x 32 keys
__global__ __launch_bounds__(256) void k_attn(
    const bf16_t* __restrict__ Qr, const bf16_t* __restrict__ Qi,
    const bf16_t* __restrict__ Kr, const bf16_t* __restrict__ Ki,
    const bf16_t* __restrict__ Vtr, const bf16_t* __restrict__ Vti,
    const int* __restrict__ mask,
    bf16_t* __restrict__ Or, bf16_t* __restrict__ Oi) {
  __shared__ bf16_t kbr[2][32 * KSTR];
  __shared__ bf16_t kbi[2][32 * KSTR];
  __shared__ bf16_t vbr[2][64 * VSTR];
  __shared__ bf16_t vbi[2][64 * VSTR];
  __shared__ float pbuf[8][16 * 36];

  const int wave = threadIdx.x >> 5;
  const int lane = threadIdx.x & 31;
  const bool hi = lane >= 16;
  const int l16 = lane & 15;

  const int rb = blockIdx.x & 15;              // N/128 row blocks
  const int h  = (blockIdx.x >> 4) & 15;
  const int b  = blockIdx.x >> 8;
  const int q0 = rb * 128 + wave * 16;
  const int hoff = h * HD_;
  const int bN = b * N_;
  const size_t vbase = (size_t)(b * H_ + h) * HD_ * (size_t)N_;

  // staging chunk coordinates for this thread (uniform category per step j)
  const int kkey = threadIdx.x >> 3, kofs = (threadIdx.x & 7) * 8;   // K: 32x128B
  const int vhd  = threadIdx.x >> 2, vofs = (threadIdx.x & 3) * 8;   // V: 64x64B

#define STAGE(nb, kb)                                                          \
  do {                                                                         \
    const bf16_t* gk = Kr + (size_t)(bN + (kb) + kkey) * D_ + hoff + kofs;     \
    const bf16_t* gi = Ki + (size_t)(bN + (kb) + kkey) * D_ + hoff + kofs;     \
    const bf16_t* gv = Vtr + vbase + (size_t)vhd * N_ + (kb) + vofs;           \
    const bf16_t* gw = Vti + vbase + (size_t)vhd * N_ + (kb) + vofs;           \
    unsigned lk = (unsigned)(uintptr_t)&kbr[nb][kkey * KSTR + kofs];           \
    unsigned li = (unsigned)(uintptr_t)&kbi[nb][kkey * KSTR + kofs];           \
    unsigned lv = (unsigned)(uintptr_t)&vbr[nb][vhd * VSTR + vofs];            \
    unsigned lw = (unsigned)(uintptr_t)&vbi[nb][vhd * VSTR + vofs];            \
    asm volatile("global_load_async_to_lds_b128 %0, %1, off"                   \
                 :: "v"(lk), "v"((unsigned long long)(uintptr_t)gk) : "memory"); \
    asm volatile("global_load_async_to_lds_b128 %0, %1, off"                   \
                 :: "v"(li), "v"((unsigned long long)(uintptr_t)gi) : "memory"); \
    asm volatile("global_load_async_to_lds_b128 %0, %1, off"                   \
                 :: "v"(lv), "v"((unsigned long long)(uintptr_t)gv) : "memory"); \
    asm volatile("global_load_async_to_lds_b128 %0, %1, off"                   \
                 :: "v"(lw), "v"((unsigned long long)(uintptr_t)gw) : "memory"); \
  } while (0)

  // Q tiles (global, loaded once)
  const size_t qrow = (size_t)(bN + q0 + l16) * D_ + hoff;
  const v16bf Aqr0 = load_A(Qr + qrow, hi), Aqr1 = load_A(Qr + qrow + 32, hi);
  const v16bf Aqi0 = load_A(Qi + qrow, hi), Aqi1 = load_A(Qi + qrow + 32, hi);

  float mrow[8], lrow[8];
  v8f accO[8];
#pragma unroll
  for (int r = 0; r < 8; ++r) { mrow[r] = -3.0e38f; lrow[r] = 0.f; }
#pragma unroll
  for (int c = 0; c < 8; ++c) accO[c] = vzero();

  float* pb = &pbuf[wave][0];
  const int* maskb = mask + bN;

  STAGE(0, 0);                                  // prologue
  for (int kb = 0; kb < N_; kb += 32) {
    const int cur = (kb >> 5) & 1;
    STAGE(cur ^ 1, (kb + 32) & (N_ - 1));       // prefetch next block
    asm volatile("s_wait_asynccnt 0x4" ::: "memory");   // cur block landed
    __syncthreads();                            // all waves' staging visible

    v8f S[2];
#pragma unroll
    for (int t = 0; t < 2; ++t) {
      const bf16_t* krow = &kbr[cur][(t * 16 + l16) * KSTR];
      const bf16_t* irow = &kbi[cur][(t * 16 + l16) * KSTR];
      v8f Sa = vzero(), Sb = vzero();
      Sa = wmma_bf16(Aqr0, load_B(krow, hi), Sa);
      Sa = wmma_bf16(Aqr1, load_B(krow + 32, hi), Sa);
      Sb = wmma_bf16(Aqi0, load_B(irow, hi), Sb);
      Sb = wmma_bf16(Aqi1, load_B(irow + 32, hi), Sb);
      v8f Sv = Sa - Sb;
      if (maskb[kb + t * 16 + l16] == 0) {
#pragma unroll
        for (int r = 0; r < 8; ++r) Sv[r] = -1.0e9f;
      }
      S[t] = Sv;
    }

    // online softmax per query row (C-layout: VGPR r -> row r / r+8)
    const int qloc = hi ? 8 : 0;
#pragma unroll
    for (int r = 0; r < 8; ++r) {
      float v = fmaxf(S[0][r], S[1][r]);
      v = fmaxf(v, __shfl_xor(v, 1, 32));
      v = fmaxf(v, __shfl_xor(v, 2, 32));
      v = fmaxf(v, __shfl_xor(v, 4, 32));
      v = fmaxf(v, __shfl_xor(v, 8, 32));
      const float mnew = fmaxf(mrow[r], v);
      const float corr = __builtin_exp2f((mrow[r] - mnew) * LOG2E_);
      const float p0 = __builtin_exp2f((S[0][r] - mnew) * LOG2E_);
      const float p1 = __builtin_exp2f((S[1][r] - mnew) * LOG2E_);
      float rs = p0 + p1;
      rs += __shfl_xor(rs, 1, 32);
      rs += __shfl_xor(rs, 2, 32);
      rs += __shfl_xor(rs, 4, 32);
      rs += __shfl_xor(rs, 8, 32);
      lrow[r] = lrow[r] * corr + rs;
      mrow[r] = mnew;
      pb[(r + qloc) * 36 + l16]      = p0;
      pb[(r + qloc) * 36 + l16 + 16] = p1;
#pragma unroll
      for (int c = 0; c < 8; ++c) accO[c][r] *= corr;
    }

    asm volatile("s_wait_dscnt 0x0" ::: "memory");   // P tile visible in LDS

    // reload P as a 16x32 A-matrix tile (bf16)
    v16bf Ap;
    {
      const float* rowp = pb + l16 * 36 + (hi ? 8 : 0);
      float4 f0 = *reinterpret_cast<const float4*>(rowp);
      float4 f1 = *reinterpret_cast<const float4*>(rowp + 4);
      float4 f2 = *reinterpret_cast<const float4*>(rowp + 16);
      float4 f3 = *reinterpret_cast<const float4*>(rowp + 20);
      Ap[0]=(bf16_t)f0.x; Ap[1]=(bf16_t)f0.y; Ap[2]=(bf16_t)f0.z; Ap[3]=(bf16_t)f0.w;
      Ap[4]=(bf16_t)f1.x; Ap[5]=(bf16_t)f1.y; Ap[6]=(bf16_t)f1.z; Ap[7]=(bf16_t)f1.w;
      Ap[8]=(bf16_t)f2.x; Ap[9]=(bf16_t)f2.y; Ap[10]=(bf16_t)f2.z; Ap[11]=(bf16_t)f2.w;
      Ap[12]=(bf16_t)f3.x; Ap[13]=(bf16_t)f3.y; Ap[14]=(bf16_t)f3.z; Ap[15]=(bf16_t)f3.w;
    }

    // accO += P @ V  (V staged transposed: contiguous along keys)
#pragma unroll
    for (int c = 0; c < 4; ++c) {
      const bf16_t* vrow = &vbr[cur][(c * 16 + l16) * VSTR];
      const bf16_t* wrow = &vbi[cur][(c * 16 + l16) * VSTR];
      accO[c]     = wmma_bf16(Ap, load_B(vrow, hi), accO[c]);
      accO[4 + c] = wmma_bf16(Ap, load_B(wrow, hi), accO[4 + c]);
    }

    __syncthreads();                 // everyone done reading cur before restage
  }

  // normalize and store attention output
#pragma unroll
  for (int r = 0; r < 8; ++r) {
    const float inv = 1.0f / lrow[r];
    const int n = q0 + r + (hi ? 8 : 0);
    const size_t rowo = (size_t)(bN + n) * D_ + hoff;
#pragma unroll
    for (int c = 0; c < 4; ++c) {
      const size_t o = rowo + c * 16 + l16;
      Or[o] = (bf16_t)(accO[c][r] * inv);
      Oi[o] = (bf16_t)(accO[4 + c][r] * inv);
    }
  }
#undef STAGE
}

// ---- output projection + residual ----------------------------------------
// yR = Zr + Or@Wor^T - Oi@Woi^T + bor ; yI = Zi + Or@Woi^T + Oi@Wor^T + boi
__global__ __launch_bounds__(256) void k_oproj(
    const bf16_t* __restrict__ Or, const bf16_t* __restrict__ Oi,
    const bf16_t* __restrict__ wor, const bf16_t* __restrict__ woi,
    const float* __restrict__ bor, const float* __restrict__ boi,
    const float* __restrict__ Zr32, const float* __restrict__ Zi32,
    float* __restrict__ out) {
  const int wave = (blockIdx.x * blockDim.x + threadIdx.x) >> 5;
  const int lane = threadIdx.x & 31;
  const bool hi = lane >= 16;
  const int l16 = lane & 15;
  const int rowT = wave & 255, colT = wave >> 8;
  const int n0 = rowT * 16, j0 = colT * 32;

  const size_t arow  = (size_t)(n0 + l16) * D_;
  const size_t brow0 = (size_t)(j0 + l16) * D_;
  const size_t brow1 = (size_t)(j0 + 16 + l16) * D_;

  v8f c1[2], c2[2], c3[2], c4[2];
#pragma unroll
  for (int u = 0; u < 2; ++u) { c1[u]=vzero(); c2[u]=vzero(); c3[u]=vzero(); c4[u]=vzero(); }

  for (int k0 = 0; k0 < D_; k0 += 32) {
    v16bf aor = load_A(Or + arow + k0, hi);
    v16bf aoi = load_A(Oi + arow + k0, hi);
#pragma unroll
    for (int u = 0; u < 2; ++u) {
      const size_t brow = u ? brow1 : brow0;
      v16bf bwr = load_B(wor + brow + k0, hi);
      v16bf bwi = load_B(woi + brow + k0, hi);
      c1[u] = wmma_bf16(aor, bwr, c1[u]);
      c2[u] = wmma_bf16(aoi, bwi, c2[u]);
      c3[u] = wmma_bf16(aor, bwi, c3[u]);
      c4[u] = wmma_bf16(aoi, bwr, c4[u]);
    }
  }

  const size_t S = (size_t)BN_ * D_;
#pragma unroll
  for (int u = 0; u < 2; ++u) {
    const int d = j0 + u * 16 + l16;
    const float br = bor[d], bi = boi[d];
    const v8f accR = c1[u] - c2[u];
    const v8f accI = c3[u] + c4[u];
#pragma unroll
    for (int r = 0; r < 8; ++r) {
      const int n = n0 + r + (hi ? 8 : 0);
      const size_t idx = (size_t)n * D_ + d;
      out[idx]     = accR[r] + br + Zr32[idx];
      out[S + idx] = accI[r] + bi + Zi32[idx];
    }
  }
}

// ---------------------------------------------------------------------------
extern "C" void kernel_launch(void* const* d_in, const int* in_sizes, int n_in,
                              void* d_out, int out_size, void* d_ws, size_t ws_size,
                              hipStream_t stream) {
  (void)in_sizes; (void)n_in; (void)out_size; (void)ws_size;
  const float* Zr32 = (const float*)d_in[0];
  const float* Zi32 = (const float*)d_in[1];
  const float* wq_r = (const float*)d_in[2];
  const float* wq_i = (const float*)d_in[3];
  const float* bq_r = (const float*)d_in[4];
  const float* bq_i = (const float*)d_in[5];
  const float* wk_r = (const float*)d_in[6];
  const float* wk_i = (const float*)d_in[7];
  const float* bk_r = (const float*)d_in[8];
  const float* bk_i = (const float*)d_in[9];
  const float* wv_r = (const float*)d_in[10];
  const float* wv_i = (const float*)d_in[11];
  const float* bv_r = (const float*)d_in[12];
  const float* bv_i = (const float*)d_in[13];
  const float* wo_r = (const float*)d_in[14];
  const float* wo_i = (const float*)d_in[15];
  const float* bo_r = (const float*)d_in[16];
  const float* bo_i = (const float*)d_in[17];
  const int*   mask = (const int*)d_in[18];

  bf16_t* w = (bf16_t*)d_ws;
  bf16_t* Zr  = w + 0 * S_;
  bf16_t* Zi  = w + 1 * S_;
  bf16_t* Qr  = w + 2 * S_;
  bf16_t* Qi  = w + 3 * S_;
  bf16_t* Kr  = w + 4 * S_;
  bf16_t* Ki  = w + 5 * S_;
  bf16_t* Vtr = w + 6 * S_;
  bf16_t* Vti = w + 7 * S_;
  bf16_t* Or  = w + 8 * S_;
  bf16_t* Oi  = w + 9 * S_;
  bf16_t* WB  = w + 10 * S_;
  bf16_t* wqr = WB + 0 * W_; bf16_t* wqi = WB + 1 * W_;
  bf16_t* wkr = WB + 2 * W_; bf16_t* wki = WB + 3 * W_;
  bf16_t* wvr = WB + 4 * W_; bf16_t* wvi = WB + 5 * W_;
  bf16_t* worb= WB + 6 * W_; bf16_t* woib= WB + 7 * W_;

  // ---- stage 0: bf16 conversions ----
  k_convert<<<1024, 256, 0, stream>>>(Zr32, Zr,  (int)S_);
  k_convert<<<1024, 256, 0, stream>>>(Zi32, Zi,  (int)S_);
  k_convert<<<512, 256, 0, stream>>>(wq_r, wqr,  (int)W_);
  k_convert<<<512, 256, 0, stream>>>(wq_i, wqi,  (int)W_);
  k_convert<<<512, 256, 0, stream>>>(wk_r, wkr,  (int)W_);
  k_convert<<<512, 256, 0, stream>>>(wk_i, wki,  (int)W_);
  k_convert<<<512, 256, 0, stream>>>(wv_r, wvr,  (int)W_);
  k_convert<<<512, 256, 0, stream>>>(wv_i, wvi,  (int)W_);
  k_convert<<<512, 256, 0, stream>>>(wo_r, worb, (int)W_);
  k_convert<<<512, 256, 0, stream>>>(wo_i, woib, (int)W_);

  // ---- stage 1: fused QKV projection (3*256*32 = 24576 waves) ----
  k_qkv<<<3072, 256, 0, stream>>>(Zr, Zi,
                                  wqr, wqi, wkr, wki, wvr, wvi,
                                  bq_r, bq_i, bk_r, bk_i, bv_r, bv_i,
                                  Qr, Qi, Kr, Ki, Vtr, Vti);

  // ---- stage 2: flash attention (B*H*(N/128) = 512 blocks) ----
  k_attn<<<512, 256, 0, stream>>>(Qr, Qi, Kr, Ki, Vtr, Vti, mask, Or, Oi);

  // ---- stage 3: O projection + residual (256*32 = 8192 waves) ----
  k_oproj<<<1024, 256, 0, stream>>>(Or, Oi, worb, woib,
                                    bo_r, bo_i, Zr32, Zi32,
                                    (float*)d_out);
}
